// spec_ln_63556926046325
// MI455X (gfx1250) — matrix operand order repaired and checked by reading, not compiled, not run on hardware
//
#include <hip/hip_runtime.h>
#include <stdint.h>

#define NN 50000
#define NPAD 50016
#define KK 1024
#define EE 200000
#define ESPEC 16384
#define NEGS 0.01f
#define EPSL 1e-5f

typedef __attribute__((ext_vector_type(16))) __bf16 v16bf;
typedef __attribute__((ext_vector_type(4)))  __bf16 v4bf;
typedef __attribute__((ext_vector_type(8)))  float  v8f;
typedef __attribute__((ext_vector_type(4)))  float  v4f;
typedef __attribute__((ext_vector_type(4)))  unsigned int v4u;

union ABu { v16bf v; v4u q[2]; unsigned short s[16]; v4bf g[4]; };

__device__ inline unsigned short f2bf(float f) {
    unsigned int u = __float_as_uint(f);
    unsigned int r = u + 0x7FFFu + ((u >> 16) & 1u);
    return (unsigned short)(r >> 16);
}
__device__ inline float lrelu(float v) { return v >= 0.f ? v : NEGS * v; }
__device__ inline unsigned int encF(float f) {
    unsigned int u = __float_as_uint(f);
    return (u & 0x80000000u) ? ~u : (u | 0x80000000u);
}
__device__ inline float decF(unsigned int e) {
    unsigned int u = (e & 0x80000000u) ? (e ^ 0x80000000u) : ~e;
    return __uint_as_float(u);
}
// packed-B layout matching wmma bf16 B-frag: lane = ((ki/8)%2)*16 + (c%16),
// element = (ki/16)*8 + ki%8, tile index = ktile*nT + c/16, 16 halves contiguous per lane.
__device__ inline long packIdx(int k, int c, int nT) {
    long kt = k >> 5; int ki = k & 31;
    int ct = c >> 4, ci = c & 15;
    int lane = (((ki >> 3) & 1) << 4) | ci;
    return (((kt * nT + ct) * 32 + lane) << 4) + (((ki >> 4) << 3) | (ki & 7));
}
__device__ inline float ln64(float v, int j, float* red, const float* g, const float* be) {
    __syncthreads();
    red[j] = v; __syncthreads();
    for (int s = 32; s > 0; s >>= 1) { if (j < s) red[j] += red[j + s]; __syncthreads(); }
    float mu = red[0] * (1.f / 64.f); __syncthreads();
    float d = v - mu; red[j] = d * d; __syncthreads();
    for (int s = 32; s > 0; s >>= 1) { if (j < s) red[j] += red[j + s]; __syncthreads(); }
    float vr = red[0] * (1.f / 64.f); __syncthreads();
    return d * rsqrtf(vr + EPSL) * g[j] + be[j];
}

// ---------------- utility kernels ----------------
__global__ void k_fill_u32(unsigned int* p, unsigned int v, long n) {
    long i = (long)blockIdx.x * blockDim.x + threadIdx.x;
    if (i < n) p[i] = v;
}
__global__ void k_u_to_bf(const float* __restrict__ u, unsigned short* __restrict__ ub) {
    long i = (long)blockIdx.x * blockDim.x + threadIdx.x;
    if (i >= (long)NPAD * KK) return;
    long n = i / KK;
    ub[i] = (n < NN) ? f2bf(u[i]) : (unsigned short)0;
}
__global__ void k_u_transpose(const float* __restrict__ u, unsigned short* __restrict__ ut) {
    __shared__ float t[32][33];
    int bn = blockIdx.x * 32, bm = blockIdx.y * 32;
    int tx = threadIdx.x, ty = threadIdx.y;   // 32 x 8
    for (int r = 0; r < 4; r++) {
        int n = bn + ty + r * 8;
        t[ty + r * 8][tx] = (n < NN) ? u[(long)n * KK + bm + tx] : 0.f;
    }
    __syncthreads();
    for (int r = 0; r < 4; r++) {
        int m = bm + ty + r * 8;
        ut[(long)m * NPAD + bn + tx] = f2bf(t[tx][ty + r * 8]);
    }
}
__global__ void k_pack_w(const float* __restrict__ W, unsigned short* __restrict__ Bp,
                         int Kd, int Cout) {
    int i = blockIdx.x * blockDim.x + threadIdx.x;
    if (i >= Kd * Cout) return;
    int k = i / Cout, c = i % Cout;
    Bp[packIdx(k, c, Cout >> 4)] = f2bf(W[(long)c * Kd + k]);   // B[k][c] = W[c][k]
}

// ---------------- per-row MLP encoder (2x linear+lrelu+LN), 64 thr/row ----------------
__global__ void k_encoder(const float* __restrict__ in, int Cin,
                          const float* __restrict__ W1, const float* __restrict__ b1,
                          const float* __restrict__ W2, const float* __restrict__ b2,
                          const float* __restrict__ g, const float* __restrict__ be,
                          float* __restrict__ D, int ldd,
                          unsigned short* __restrict__ P, int nT) {
    int row = blockIdx.x, j = threadIdx.x;
    __shared__ float sh[64];
    __shared__ float red[64];
    float a = b1[j];
    for (int c = 0; c < Cin; c++) a += in[(long)row * Cin + c] * W1[j * Cin + c];
    a = lrelu(a);
    float h1 = ln64(a, j, red, g, be);
    __syncthreads();
    sh[j] = h1; __syncthreads();
    float a2 = b2[j];
    for (int k = 0; k < 64; k++) a2 += sh[k] * W2[j * 64 + k];
    a2 = lrelu(a2);
    float o = ln64(a2, j, red, g, be);
    D[(long)row * ldd + j] = o;
    if (P) P[packIdx(row, j, nT)] = f2bf(o);
}

// ---------------- EdgeConv: gather-fused bf16 WMMA GEMM + atomic segment-max ----------
// K-loop split into two branch-free phases: [0,Cin) reads x_i, [Cin,2Cin) reads x_j - x_i.
__global__ void k_ec_gemm(const float* __restrict__ X, int ldx, int Cin,
                          const int* __restrict__ esrc, const int* __restrict__ edst, int E,
                          const unsigned short* __restrict__ Bp,
                          const float* __restrict__ bias,
                          unsigned int* __restrict__ Out) {
    int wid = (blockIdx.x * blockDim.x + threadIdx.x) >> 5;
    int lane = threadIdx.x & 31;
    int row0 = wid << 4;
    if (row0 >= E) return;
    int er = row0 + (lane & 15);
    if (er >= E) er = E - 1;
    int si = esrc[er], di = edst[er];
    const float* ps = X + (long)si * ldx;
    const float* pd = X + (long)di * ldx;
    int kofs = (lane >> 4) << 3;
    v8f acc[4] = {};
    int kHalf = Cin >> 5;                 // Cin is a multiple of 32 for all stages
    // phase A: features [0, Cin) = x_i
    for (int ks = 0; ks < kHalf; ++ks) {
        ABu a;
        int f0 = ks * 32 + kofs;
        a.g[0] = __builtin_convertvector(*(const v4f*)(pd + f0), v4bf);
        a.g[1] = __builtin_convertvector(*(const v4f*)(pd + f0 + 4), v4bf);
        a.g[2] = __builtin_convertvector(*(const v4f*)(pd + f0 + 16), v4bf);
        a.g[3] = __builtin_convertvector(*(const v4f*)(pd + f0 + 20), v4bf);
        long bbase = (((long)ks * 4) * 32 + lane) << 4;
#pragma unroll
        for (int t = 0; t < 4; t++) {
            ABu b;
            b.q[0] = *(const v4u*)(Bp + bbase + t * 512);
            b.q[1] = *(const v4u*)(Bp + bbase + t * 512 + 8);
            acc[t] = __builtin_amdgcn_wmma_f32_16x16x32_bf16(false, a.v, false, b.v,
                                                             (short)0, acc[t], false, false);
        }
    }
    // phase B: features [Cin, 2Cin) = x_j - x_i
    for (int ks2 = 0; ks2 < kHalf; ++ks2) {
        ABu a;
        int ks = kHalf + ks2;
        int f0 = ks2 * 32 + kofs;
        v4f d0 = *(const v4f*)(ps + f0)      - *(const v4f*)(pd + f0);
        v4f d1 = *(const v4f*)(ps + f0 + 4)  - *(const v4f*)(pd + f0 + 4);
        v4f d2 = *(const v4f*)(ps + f0 + 16) - *(const v4f*)(pd + f0 + 16);
        v4f d3 = *(const v4f*)(ps + f0 + 20) - *(const v4f*)(pd + f0 + 20);
        a.g[0] = __builtin_convertvector(d0, v4bf);
        a.g[1] = __builtin_convertvector(d1, v4bf);
        a.g[2] = __builtin_convertvector(d2, v4bf);
        a.g[3] = __builtin_convertvector(d3, v4bf);
        long bbase = (((long)ks * 4) * 32 + lane) << 4;
#pragma unroll
        for (int t = 0; t < 4; t++) {
            ABu b;
            b.q[0] = *(const v4u*)(Bp + bbase + t * 512);
            b.q[1] = *(const v4u*)(Bp + bbase + t * 512 + 8);
            acc[t] = __builtin_amdgcn_wmma_f32_16x16x32_bf16(false, a.v, false, b.v,
                                                             (short)0, acc[t], false, false);
        }
    }
    int col = lane & 15;
    int hi = (lane >> 4) << 3;
    int dsts[8];
#pragma unroll
    for (int r = 0; r < 8; r++) { int m = row0 + hi + r; dsts[r] = (m < E) ? edst[m] : -1; }
#pragma unroll
    for (int t = 0; t < 4; t++) {
        int c = (t << 4) + col;
        float bv = bias[c];
#pragma unroll
        for (int r = 0; r < 8; r++) {
            if (dsts[r] >= 0)
                atomicMax(&Out[(long)dsts[r] * 64 + c], encF(acc[t][r] + bv));
        }
    }
}

__global__ void k_ec_finalize(const unsigned int* __restrict__ enc,
                              const float* __restrict__ g, const float* __restrict__ be,
                              float* __restrict__ D, int ldd, int colOff,
                              unsigned short* __restrict__ P, int nT, int pColOff) {
    int row = blockIdx.x, j = threadIdx.x;
    __shared__ float red[64];
    float v = decF(enc[(long)row * 64 + j]);
    if (!isfinite(v)) v = 0.f;
    v = lrelu(v);
    float o = ln64(v, j, red, g, be);
    D[(long)row * ldd + colOff + j] = o;
    if (P) P[packIdx(row, pColOff + j, nT)] = f2bf(o);
}

// ---------------- u @ S : bf16 A rows (pitch 1024), packed B ----------------
__global__ void k_gemm_uS(const unsigned short* __restrict__ A,
                          const unsigned short* __restrict__ Bp, int nTB,
                          int nRowTiles, int Mstore, int kSteps, int colGroups,
                          float* __restrict__ D, int ldd, int colOff,
                          unsigned short* __restrict__ P, int nTP, int pColOff) {
    int wid = (blockIdx.x * blockDim.x + threadIdx.x) >> 5;
    int lane = threadIdx.x & 31;
    int rt = wid % nRowTiles;
    int cg = wid / nRowTiles;
    if (cg >= colGroups) return;
    int row0 = rt << 4;
    int arow = row0 + (lane & 15);
    int kofs = (lane >> 4) << 3;
    v8f acc[4] = {};
    const unsigned short* Arow = A + (long)arow * 1024 + kofs;
    for (int ks = 0; ks < kSteps; ++ks) {
        __builtin_prefetch(Arow + (ks + 8) * 32, 0, 0);   // stream-ahead: global_prefetch_b8
        ABu a;
        a.q[0] = *(const v4u*)(Arow + ks * 32);
        a.q[1] = *(const v4u*)(Arow + ks * 32 + 16);
        long bbase = (((long)ks * nTB + (cg << 2)) * 32 + lane) << 4;
#pragma unroll
        for (int t = 0; t < 4; t++) {
            ABu b;
            b.q[0] = *(const v4u*)(Bp + bbase + t * 512);
            b.q[1] = *(const v4u*)(Bp + bbase + t * 512 + 8);
            acc[t] = __builtin_amdgcn_wmma_f32_16x16x32_bf16(false, a.v, false, b.v,
                                                             (short)0, acc[t], false, false);
        }
    }
    int col = lane & 15;
    int hi = (lane >> 4) << 3;
#pragma unroll
    for (int t = 0; t < 4; t++) {
        int cc = (cg << 6) + (t << 4) + col;
#pragma unroll
        for (int r = 0; r < 8; r++) {
            int m = row0 + hi + r;
            if (m < Mstore) {
                float v = acc[t][r];
                D[(long)m * ldd + colOff + cc] = v;
                if (P) P[packIdx(m, pColOff + cc, nTP)] = f2bf(v);
            }
        }
    }
}

// ---------------- u^T @ X : split-K (32 slices) into partial buffer ----------------
__global__ void k_gemm_uT(const unsigned short* __restrict__ A,   // [1024, NPAD] bf16
                          const unsigned short* __restrict__ Bp, int nTB,
                          int colGroups, int C,
                          float* __restrict__ Part) {
    int wid = (blockIdx.x * blockDim.x + threadIdx.x) >> 5;
    int lane = threadIdx.x & 31;
    int rt = wid & 63;
    int tmp = wid >> 6;
    int cg = tmp % colGroups;
    int sl = tmp / colGroups;
    if (sl >= 32) return;
    const int kSteps = NPAD / 32;                  // 1563
    const int kPer = (kSteps + 31) / 32;           // 49
    int k0 = sl * kPer;
    int k1 = (k0 + kPer < kSteps) ? (k0 + kPer) : kSteps;
    int row0 = rt << 4;
    int arow = row0 + (lane & 15);
    int kofs = (lane >> 4) << 3;
    v8f acc[4] = {};
    const unsigned short* Arow = A + (long)arow * NPAD + kofs;
    for (int ks = k0; ks < k1; ++ks) {
        __builtin_prefetch(Arow + (ks + 8) * 32, 0, 0);
        ABu a;
        a.q[0] = *(const v4u*)(Arow + ks * 32);
        a.q[1] = *(const v4u*)(Arow + ks * 32 + 16);
        long bbase = (((long)ks * nTB + (cg << 2)) * 32 + lane) << 4;
#pragma unroll
        for (int t = 0; t < 4; t++) {
            ABu b;
            b.q[0] = *(const v4u*)(Bp + bbase + t * 512);
            b.q[1] = *(const v4u*)(Bp + bbase + t * 512 + 8);
            acc[t] = __builtin_amdgcn_wmma_f32_16x16x32_bf16(false, a.v, false, b.v,
                                                             (short)0, acc[t], false, false);
        }
    }
    int col = lane & 15;
    int hi = (lane >> 4) << 3;
    float* P0 = Part + (long)sl * 1024 * C;
#pragma unroll
    for (int t = 0; t < 4; t++) {
        int cc = (cg << 6) + (t << 4) + col;
#pragma unroll
        for (int r = 0; r < 8; r++) {
            int m = row0 + hi + r;
            P0[(long)m * C + cc] = acc[t][r];
        }
    }
}

__global__ void k_reduce_part(const float* __restrict__ Part, int C,
                              float* __restrict__ D, int ldd, int colOff,
                              unsigned short* __restrict__ P, int nTP, int pColOff) {
    int i = blockIdx.x * blockDim.x + threadIdx.x;
    if (i >= 1024 * C) return;
    int m = i / C, c = i % C;
    float s = 0.f;
    for (int sl = 0; sl < 32; ++sl) s += Part[((long)sl * 1024 + m) * C + c];
    D[(long)m * ldd + colOff + c] = s;
    if (P) P[packIdx(m, pColOff + c, nTP)] = f2bf(s);
}

// ---------------- final dense d1 GEMM over cat(Xbuf, X4buf), two branch-free phases ----
__global__ void k_d1_gemm(const float* __restrict__ Xb, const float* __restrict__ X4,
                          const unsigned short* __restrict__ Bp,
                          const float* __restrict__ bias, float* __restrict__ H) {
    int wid = (blockIdx.x * blockDim.x + threadIdx.x) >> 5;
    int lane = threadIdx.x & 31;
    int row0 = wid << 4;
    if (row0 >= NN) return;
    int ar = row0 + (lane & 15);
    if (ar >= NN) ar = NN - 1;
    const float* p1 = Xb + (long)ar * 640;
    const float* p2 = X4 + (long)ar * 320;
    int kofs = (lane >> 4) << 3;
    v8f acc[4] = {};
    for (int ks = 0; ks < 20; ++ks) {           // features [0,640) from Xb
        ABu a;
        int f0 = ks * 32 + kofs;
        a.g[0] = __builtin_convertvector(*(const v4f*)(p1 + f0), v4bf);
        a.g[1] = __builtin_convertvector(*(const v4f*)(p1 + f0 + 4), v4bf);
        a.g[2] = __builtin_convertvector(*(const v4f*)(p1 + f0 + 16), v4bf);
        a.g[3] = __builtin_convertvector(*(const v4f*)(p1 + f0 + 20), v4bf);
        long bbase = (((long)ks * 4) * 32 + lane) << 4;
#pragma unroll
        for (int t = 0; t < 4; t++) {
            ABu b;
            b.q[0] = *(const v4u*)(Bp + bbase + t * 512);
            b.q[1] = *(const v4u*)(Bp + bbase + t * 512 + 8);
            acc[t] = __builtin_amdgcn_wmma_f32_16x16x32_bf16(false, a.v, false, b.v,
                                                             (short)0, acc[t], false, false);
        }
    }
    for (int ks2 = 0; ks2 < 10; ++ks2) {        // features [640,960) from X4
        ABu a;
        int ks = 20 + ks2;
        int f0 = ks2 * 32 + kofs;
        a.g[0] = __builtin_convertvector(*(const v4f*)(p2 + f0), v4bf);
        a.g[1] = __builtin_convertvector(*(const v4f*)(p2 + f0 + 4), v4bf);
        a.g[2] = __builtin_convertvector(*(const v4f*)(p2 + f0 + 16), v4bf);
        a.g[3] = __builtin_convertvector(*(const v4f*)(p2 + f0 + 20), v4bf);
        long bbase = (((long)ks * 4) * 32 + lane) << 4;
#pragma unroll
        for (int t = 0; t < 4; t++) {
            ABu b;
            b.q[0] = *(const v4u*)(Bp + bbase + t * 512);
            b.q[1] = *(const v4u*)(Bp + bbase + t * 512 + 8);
            acc[t] = __builtin_amdgcn_wmma_f32_16x16x32_bf16(false, a.v, false, b.v,
                                                             (short)0, acc[t], false, false);
        }
    }
    int col = lane & 15;
    int hi = (lane >> 4) << 3;
#pragma unroll
    for (int t = 0; t < 4; t++) {
        int c = (t << 4) + col;
        float bv = bias[c];
#pragma unroll
        for (int r = 0; r < 8; r++) {
            int m = row0 + hi + r;
            if (m < NN) H[(long)m * 64 + c] = acc[t][r] + bv;
        }
    }
}

__global__ void k_final(const float* __restrict__ H,
                        const float* __restrict__ g, const float* __restrict__ be,
                        const float* __restrict__ d2w, const float* __restrict__ d2b,
                        float* __restrict__ out) {
    int row = blockIdx.x, j = threadIdx.x;
    __shared__ float red[64];
    __shared__ float sh[64];
    float v = lrelu(H[(long)row * 64 + j]);
    float o = ln64(v, j, red, g, be);
    __syncthreads();
    sh[j] = o; __syncthreads();
    if (j < 32) {
        float s = d2b[j];
        for (int k = 0; k < 64; k++) s += sh[k] * d2w[j * 64 + k];
        out[(long)row * 32 + j] = s;
    }
}

// =============================================================================
extern "C" void kernel_launch(void* const* d_in, const int* in_sizes, int n_in,
                              void* d_out, int out_size, void* d_ws, size_t ws_size,
                              hipStream_t stream) {
    const float* x    = (const float*)d_in[0];
    const float* u    = (const float*)d_in[1];
    const float* e    = (const float*)d_in[2];
    const int*   ei   = (const int*)d_in[3];
    const int*   se   = (const int*)d_in[4];
    const float* se1_w = (const float*)d_in[5];  const float* se1_b = (const float*)d_in[6];
    const float* se2_w = (const float*)d_in[7];  const float* se2_b = (const float*)d_in[8];
    const float* sp1_w = (const float*)d_in[9];  const float* sp1_b = (const float*)d_in[10];
    const float* sp2_w = (const float*)d_in[11]; const float* sp2_b = (const float*)d_in[12];
    const float* c1_w = (const float*)d_in[13];  const float* c1_b = (const float*)d_in[14];
    const float* c2_w = (const float*)d_in[15];  const float* c2_b = (const float*)d_in[16];
    const float* c3_w = (const float*)d_in[17];  const float* c3_b = (const float*)d_in[18];
    const float* c5_w = (const float*)d_in[19];  const float* c5_b = (const float*)d_in[20];
    const float* c6_w = (const float*)d_in[21];  const float* c6_b = (const float*)d_in[22];
    const float* c7_w = (const float*)d_in[23];  const float* c7_b = (const float*)d_in[24];
    const float* c8_w = (const float*)d_in[25];  const float* c8_b = (const float*)d_in[26];
    const float* d1_w = (const float*)d_in[27];  const float* d1_b = (const float*)d_in[28];
    const float* d2_w = (const float*)d_in[29];  const float* d2_b = (const float*)d_in[30];
    const float* ln_g = (const float*)d_in[31];  const float* ln_b = (const float*)d_in[32];
    float* out = (float*)d_out;
    (void)in_sizes; (void)n_in; (void)out_size; (void)ws_size;

    char* base = (char*)d_ws;
    size_t off = 0;
    auto alloc = [&](size_t bytes) -> char* {
        char* p = base + off;
        off = (off + bytes + 255) & ~(size_t)255;
        return p;
    };
    unsigned short* u_bf  = (unsigned short*)alloc((size_t)NPAD * KK * 2);
    unsigned short* ut_bf = (unsigned short*)alloc((size_t)KK * NPAD * 2);
    float* Xb = (float*)alloc((size_t)NN * 640 * 4);
    float* X4 = (float*)alloc((size_t)NN * 320 * 4);
    float* Sb = (float*)alloc((size_t)KK * 384 * 4);
    float* S3 = (float*)alloc((size_t)KK * 256 * 4);
    unsigned short* pX0 = (unsigned short*)alloc((size_t)NPAD * 64 * 2);
    unsigned short* pX1 = (unsigned short*)alloc((size_t)NPAD * 128 * 2);
    unsigned short* pX2 = (unsigned short*)alloc((size_t)NPAD * 192 * 2);
    unsigned short* pS0 = (unsigned short*)alloc((size_t)KK * 64 * 2);
    unsigned short* pS1 = (unsigned short*)alloc((size_t)KK * 128 * 2);
    unsigned short* pS2 = (unsigned short*)alloc((size_t)KK * 192 * 2);
    unsigned short* pS3 = (unsigned short*)alloc((size_t)KK * 256 * 2);
    unsigned short* pWc1 = (unsigned short*)alloc((size_t)128 * 64 * 2);
    unsigned short* pWc2 = (unsigned short*)alloc((size_t)384 * 64 * 2);
    unsigned short* pWc3 = (unsigned short*)alloc((size_t)768 * 64 * 2);
    unsigned short* pWc5 = (unsigned short*)alloc((size_t)128 * 64 * 2);
    unsigned short* pWc6 = (unsigned short*)alloc((size_t)384 * 64 * 2);
    unsigned short* pWc7 = (unsigned short*)alloc((size_t)768 * 64 * 2);
    unsigned short* pWc8 = (unsigned short*)alloc((size_t)1280 * 64 * 2);
    unsigned short* pWd1 = (unsigned short*)alloc((size_t)960 * 64 * 2);
    unsigned int* ecb = (unsigned int*)alloc((size_t)NN * 64 * 4);
    float* Part = (float*)alloc((size_t)32 * KK * 192 * 4);
    float* Hpre = (float*)alloc((size_t)NN * 64 * 4);

    const int* esrc = ei;       const int* edst = ei + EE;
    const int* ssrc = se;       const int* sdst = se + ESPEC;

    dim3 b256(256), b128(128), b64(64);
    auto gFor   = [](long n, int bs) { return dim3((unsigned)((n + bs - 1) / bs)); };
    auto gWaves = [](long w)         { return dim3((unsigned)((w + 3) / 4)); };
    const unsigned int ENC_NEG_INF = 0x007FFFFFu;

    // --- one-time u conversions + weight packing + pad zeroing ---
    k_u_to_bf<<<gFor((long)NPAD * KK, 256), b256, 0, stream>>>(u, u_bf);
    k_u_transpose<<<dim3(NPAD / 32, KK / 32), dim3(32, 8), 0, stream>>>(u, ut_bf);
    k_pack_w<<<gFor(128 * 64, 256), b256, 0, stream>>>(c1_w, pWc1, 128, 64);
    k_pack_w<<<gFor(384 * 64, 256), b256, 0, stream>>>(c2_w, pWc2, 384, 64);
    k_pack_w<<<gFor(768 * 64, 256), b256, 0, stream>>>(c3_w, pWc3, 768, 64);
    k_pack_w<<<gFor(128 * 64, 256), b256, 0, stream>>>(c5_w, pWc5, 128, 64);
    k_pack_w<<<gFor(384 * 64, 256), b256, 0, stream>>>(c6_w, pWc6, 384, 64);
    k_pack_w<<<gFor(768 * 64, 256), b256, 0, stream>>>(c7_w, pWc7, 768, 64);
    k_pack_w<<<gFor(1280 * 64, 256), b256, 0, stream>>>(c8_w, pWc8, 1280, 64);
    k_pack_w<<<gFor(960 * 64, 256), b256, 0, stream>>>(d1_w, pWd1, 960, 64);
    k_fill_u32<<<gFor((long)NPAD * 64 / 2, 256), b256, 0, stream>>>((unsigned int*)pX0, 0u, (long)NPAD * 64 / 2);
    k_fill_u32<<<gFor((long)NPAD * 128 / 2, 256), b256, 0, stream>>>((unsigned int*)pX1, 0u, (long)NPAD * 128 / 2);
    k_fill_u32<<<gFor((long)NPAD * 192 / 2, 256), b256, 0, stream>>>((unsigned int*)pX2, 0u, (long)NPAD * 192 / 2);

    // --- encoders: s0, x0 ---
    k_encoder<<<dim3(KK), b64, 0, stream>>>(e, 1, se1_w, se1_b, se2_w, se2_b, ln_g, ln_b, Sb, 384, pS0, 4);
    k_encoder<<<dim3(NN), b64, 0, stream>>>(x, 3, sp1_w, sp1_b, sp2_w, sp2_b, ln_g, ln_b, Xb, 640, pX0, 4);

    // --- stage 1 ---
    k_fill_u32<<<gFor((long)KK * 64, 256), b256, 0, stream>>>(ecb, ENC_NEG_INF, (long)KK * 64);
    k_ec_gemm<<<gWaves((ESPEC + 15) / 16), b128, 0, stream>>>(Sb, 384, 64, ssrc, sdst, ESPEC, pWc1, c1_b, ecb);
    k_ec_finalize<<<dim3(KK), b64, 0, stream>>>(ecb, ln_g, ln_b, Sb, 384, 64, pS1, 8, 0);
    k_fill_u32<<<gFor((long)NN * 64, 256), b256, 0, stream>>>(ecb, ENC_NEG_INF, (long)NN * 64);
    k_ec_gemm<<<gWaves((EE + 15) / 16), b128, 0, stream>>>(Xb, 640, 64, esrc, edst, EE, pWc5, c5_b, ecb);
    k_ec_finalize<<<dim3(NN), b64, 0, stream>>>(ecb, ln_g, ln_b, Xb, 640, 64, pX1, 8, 0);
    k_gemm_uT<<<gWaves((long)64 * 1 * 32), b128, 0, stream>>>(ut_bf, pX0, 4, 1, 64, Part);
    k_reduce_part<<<gFor(KK * 64, 256), b256, 0, stream>>>(Part, 64, Sb, 384, 128, pS1, 8, 64);
    k_gemm_uS<<<gWaves((long)3126 * 1), b128, 0, stream>>>(u_bf, pS0, 4, 3126, NN, 32, 1, Xb, 640, 128, pX1, 8, 64);

    // --- stage 2 ---
    k_fill_u32<<<gFor((long)KK * 64, 256), b256, 0, stream>>>(ecb, ENC_NEG_INF, (long)KK * 64);
    k_ec_gemm<<<gWaves((ESPEC + 15) / 16), b128, 0, stream>>>(Sb, 384, 192, ssrc, sdst, ESPEC, pWc2, c2_b, ecb);
    k_ec_finalize<<<dim3(KK), b64, 0, stream>>>(ecb, ln_g, ln_b, Sb, 384, 192, pS2, 12, 0);
    k_fill_u32<<<gFor((long)NN * 64, 256), b256, 0, stream>>>(ecb, ENC_NEG_INF, (long)NN * 64);
    k_ec_gemm<<<gWaves((EE + 15) / 16), b128, 0, stream>>>(Xb, 640, 192, esrc, edst, EE, pWc6, c6_b, ecb);
    k_ec_finalize<<<dim3(NN), b64, 0, stream>>>(ecb, ln_g, ln_b, Xb, 640, 192, pX2, 12, 0);
    k_gemm_uT<<<gWaves((long)64 * 2 * 32), b128, 0, stream>>>(ut_bf, pX1, 8, 2, 128, Part);
    k_reduce_part<<<gFor(KK * 128, 256), b256, 0, stream>>>(Part, 128, Sb, 384, 256, pS2, 12, 64);
    k_gemm_uS<<<gWaves((long)3126 * 2), b128, 0, stream>>>(u_bf, pS1, 8, 3126, NN, 32, 2, Xb, 640, 256, pX2, 12, 64);

    // --- stage 3 ---
    k_fill_u32<<<gFor((long)KK * 64, 256), b256, 0, stream>>>(ecb, ENC_NEG_INF, (long)KK * 64);
    k_ec_gemm<<<gWaves((ESPEC + 15) / 16), b128, 0, stream>>>(Sb, 384, 384, ssrc, sdst, ESPEC, pWc3, c3_b, ecb);
    k_ec_finalize<<<dim3(KK), b64, 0, stream>>>(ecb, ln_g, ln_b, S3, 256, 0, pS3, 16, 0);
    k_fill_u32<<<gFor((long)NN * 64, 256), b256, 0, stream>>>(ecb, ENC_NEG_INF, (long)NN * 64);
    k_ec_gemm<<<gWaves((EE + 15) / 16), b128, 0, stream>>>(Xb, 640, 384, esrc, edst, EE, pWc7, c7_b, ecb);
    k_ec_finalize<<<dim3(NN), b64, 0, stream>>>(ecb, ln_g, ln_b, Xb, 640, 384, (unsigned short*)nullptr, 0, 0);
    k_gemm_uT<<<gWaves((long)64 * 3 * 32), b128, 0, stream>>>(ut_bf, pX2, 12, 3, 192, Part);
    k_reduce_part<<<gFor(KK * 192, 256), b256, 0, stream>>>(Part, 192, S3, 256, 64, pS3, 16, 64);
    k_gemm_uS<<<gWaves((long)3126 * 3), b128, 0, stream>>>(u_bf, pS2, 12, 3126, NN, 32, 3, Xb, 640, 448,
                                                           (unsigned short*)nullptr, 0, 0);

    // --- stage 4 ---
    k_fill_u32<<<gFor((long)NN * 64, 256), b256, 0, stream>>>(ecb, ENC_NEG_INF, (long)NN * 64);
    k_ec_gemm<<<gWaves((EE + 15) / 16), b128, 0, stream>>>(Xb, 640, 640, esrc, edst, EE, pWc8, c8_b, ecb);
    k_ec_finalize<<<dim3(NN), b64, 0, stream>>>(ecb, ln_g, ln_b, X4, 320, 0, (unsigned short*)nullptr, 0, 0);
    k_gemm_uS<<<gWaves((long)3126 * 4), b128, 0, stream>>>(u_bf, pS3, 16, 3126, NN, 32, 4, X4, 320, 64,
                                                           (unsigned short*)nullptr, 0, 0);

    // --- decoder ---
    k_d1_gemm<<<gWaves((NN + 15) / 16), b128, 0, stream>>>(Xb, X4, pWd1, d1_b, Hpre);
    k_final<<<dim3(NN), b64, 0, stream>>>(Hpre, ln_g, ln_b, d2_w, d2_b, out);
}
